// EdgeProcessor_70944269796072
// MI455X (gfx1250) — matrix operand order repaired and checked by reading, not compiled, not run on hardware
//
#include <hip/hip_runtime.h>

// ---------------- CDNA5 WMMA types ----------------
typedef __attribute__((ext_vector_type(16))) __bf16 v16bf;
typedef __attribute__((ext_vector_type(8)))  __bf16 v8bf;
typedef __attribute__((ext_vector_type(8)))  float  v8f;

union AFrag { v16bf v; v8bf h[2]; };

__device__ __forceinline__ unsigned short f2bf(float f) {
    unsigned u = __float_as_uint(f);
    u += 0x7FFFu + ((u >> 16) & 1u);   // round-to-nearest-even
    return (unsigned short)(u >> 16);
}

__device__ __forceinline__ v8f zero8() {
    v8f z;
#pragma unroll
    for (int i = 0; i < 8; ++i) z[i] = 0.f;
    return z;
}

// convert float4 -> 4 packed bf16, store 8B to LDS
__device__ __forceinline__ void st4bf(unsigned short* dst, float4 v) {
    uint2 p;
    p.x = (unsigned)f2bf(v.x) | ((unsigned)f2bf(v.y) << 16);
    p.y = (unsigned)f2bf(v.z) | ((unsigned)f2bf(v.w) << 16);
    *(uint2*)dst = p;
}

// A fragment: 16-bit A 16x32 layout. lane<16: K 0-15; lane>=16: K 16-31 (contiguous, matches B prepack).
__device__ __forceinline__ v16bf load_afrag(const unsigned short* p) {
    AFrag a;
    a.h[0] = *(const v8bf*)(p);
    a.h[1] = *(const v8bf*)(p + 8);
    return a.v;
}

// ---------------- weight prepack: f32 [K,N] -> bf16 B-fragments ----------------
// fragment (ntile, kc): 32 lanes x 16 bf16, lane-contiguous (32B per lane).
// lane maps to column n = ntile*16 + (lane&15); lanes 0-15 carry K=kc*32+0..15,
// lanes 16-31 carry K=kc*32+16..31 (each lane: 16 consecutive K values).
__global__ void prepack_w(const float* __restrict__ W, unsigned short* __restrict__ out,
                          int K, int N) {
    int tid = blockIdx.x * blockDim.x + threadIdx.x;
    int KC  = K >> 5;
    int nfr = (N >> 4) * KC;
    if (tid >= nfr * 32) return;
    int lane  = tid & 31;
    int frag  = tid >> 5;
    int kc    = frag % KC;
    int ntile = frag / KC;
    int n  = (ntile << 4) | (lane & 15);
    int kb = (kc << 5) + ((lane & 16) ? 16 : 0);
    unsigned short* o = out + ((size_t)tid << 4);
#pragma unroll
    for (int e = 0; e < 16; ++e)
        o[e] = f2bf(W[(size_t)(kb + e) * N + n]);
}

// ---------------- fused GEMM tile (one wave: 16 cols, two 16-row subtiles) ----------
template <int KC, int PITCH>
__device__ __forceinline__ void tile_gemm(const unsigned short* Asrc,
                                          const unsigned short* __restrict__ Wp,
                                          int wave, int lane, v8f& acc0, v8f& acc1) {
    const int m  = lane & 15;
    const int kb = (lane & 16) ? 16 : 0;                // A fragment K-base per half-wave
    const unsigned short* a0p = Asrc + m * PITCH + kb;
    const unsigned short* a1p = a0p + 16 * PITCH;
    const unsigned short* wp  = Wp + (((size_t)wave * KC) * 32 + lane) * 16;
#pragma unroll
    for (int kc = 0; kc < KC; ++kc) {
        v16bf b  = *(const v16bf*)(wp);                 // 32B contiguous per lane
        wp += 32 * 16;
        v16bf a0 = load_afrag(a0p + kc * 32);
        v16bf a1 = load_afrag(a1p + kc * 32);
        acc0 = __builtin_amdgcn_wmma_f32_16x16x32_bf16(false, a0, false, b, (short)0, acc0, false, false);
        acc1 = __builtin_amdgcn_wmma_f32_16x16x32_bf16(false, a1, false, b, (short)0, acc1, false, false);
    }
}

// ---------------- main fused kernel ----------------
#define D       128
#define MTILE   32
#define XPITCH  392   // 384 + 8 halves pad
#define HPITCH  136   // 128 + 8 halves pad
#define YPITCH  132   // 128 + 4 floats pad

__global__ __launch_bounds__(256) void edge_mlp_kernel(
    const float* __restrict__ sf, const float* __restrict__ rf, const float* __restrict__ ef,
    const int* __restrict__ snd, const int* __restrict__ rcv,
    const unsigned short* __restrict__ W1p, const float* __restrict__ b1,
    const unsigned short* __restrict__ W2p, const float* __restrict__ b2,
    const unsigned short* __restrict__ W3p, const float* __restrict__ b3,
    const float* __restrict__ gamma, const float* __restrict__ beta,
    float* __restrict__ outp, int E) {

    __shared__ __attribute__((aligned(16))) unsigned short Xs[MTILE * XPITCH];
    __shared__ __attribute__((aligned(16))) unsigned short H1s[MTILE * HPITCH];
    __shared__ __attribute__((aligned(16))) unsigned short H2s[MTILE * HPITCH];
    __shared__ __attribute__((aligned(16))) float Ys[MTILE * YPITCH];

    const int tid = threadIdx.x;
    const int e0  = blockIdx.x * MTILE;

    // ---- gather sender/receiver/edge rows, convert f32 -> bf16 into X tile ----
    {
        int el   = tid >> 3;          // edge-local row 0..31
        int sub  = tid & 7;           // 8 threads per edge
        int edge = e0 + el;
        if (edge >= E) edge = E - 1;  // clamp tail (rows masked at output)
        int s = snd[edge], r = rcv[edge];
        const float4* srow = (const float4*)(sf + (size_t)s * D);
        const float4* rrow = (const float4*)(rf + (size_t)r * D);
        const float4* erow = (const float4*)(ef + (size_t)edge * D);
        unsigned short* xr = Xs + el * XPITCH;
#pragma unroll
        for (int j = 0; j < 4; ++j) {
            int c4 = (sub << 2) + j;                // float4 index within 128 floats
            st4bf(xr + c4 * 4,        srow[c4]);    // cols   0..127
            st4bf(xr + 128 + c4 * 4,  rrow[c4]);    // cols 128..255
            st4bf(xr + 256 + c4 * 4,  erow[c4]);    // cols 256..383
        }
    }
    __syncthreads();

    const int wave = tid >> 5;
    const int lane = tid & 31;
    const int col  = (wave << 4) | (lane & 15);     // global hidden column of this lane
    const int rsel = (lane & 16) ? 8 : 0;           // C/D row offset per half-wave

    // ---- layer 1: [32,384] x [384,128] ----
    {
        v8f a0 = zero8(), a1 = zero8();
        tile_gemm<12, XPITCH>(Xs, W1p, wave, lane, a0, a1);
        float bias = b1[col];
#pragma unroll
        for (int r = 0; r < 8; ++r) {
            float v0 = fmaxf(a0[r] + bias, 0.f);
            float v1 = fmaxf(a1[r] + bias, 0.f);
            H1s[(rsel + r) * HPITCH + col]      = f2bf(v0);
            H1s[(16 + rsel + r) * HPITCH + col] = f2bf(v1);
        }
    }
    __syncthreads();

    // ---- layer 2: [32,128] x [128,128] ----
    {
        v8f a0 = zero8(), a1 = zero8();
        tile_gemm<4, HPITCH>(H1s, W2p, wave, lane, a0, a1);
        float bias = b2[col];
#pragma unroll
        for (int r = 0; r < 8; ++r) {
            float v0 = fmaxf(a0[r] + bias, 0.f);
            float v1 = fmaxf(a1[r] + bias, 0.f);
            H2s[(rsel + r) * HPITCH + col]      = f2bf(v0);
            H2s[(16 + rsel + r) * HPITCH + col] = f2bf(v1);
        }
    }
    __syncthreads();

    // ---- layer 3 (linear): [32,128] x [128,128] -> f32 Y tile ----
    {
        v8f a0 = zero8(), a1 = zero8();
        tile_gemm<4, HPITCH>(H2s, W3p, wave, lane, a0, a1);
        float bias = b3[col];
#pragma unroll
        for (int r = 0; r < 8; ++r) {
            Ys[(rsel + r) * YPITCH + col]      = a0[r] + bias;
            Ys[(16 + rsel + r) * YPITCH + col] = a1[r] + bias;
        }
    }
    __syncthreads();

    // ---- LayerNorm over D=128 + affine, coalesced store ----
    {
        int row  = tid >> 3;                        // 0..31
        int part = tid & 7;                         // 8 threads per row
        const float* yr = Ys + row * YPITCH + part * 16;
        float s = 0.f, s2 = 0.f;
#pragma unroll
        for (int j = 0; j < 16; ++j) { float v = yr[j]; s += v; s2 += v * v; }
#pragma unroll
        for (int m = 4; m; m >>= 1) {               // butterfly within width-8 groups
            s  += __shfl_xor(s,  m, 8);
            s2 += __shfl_xor(s2, m, 8);
        }
        float mu  = s * (1.f / 128.f);
        float var = fmaxf(s2 * (1.f / 128.f) - mu * mu, 0.f);
        float inv = rsqrtf(var + 1e-6f);
        int edge = e0 + row;
        if (edge < E) {
            float* orow = outp + (size_t)edge * D + part * 16;
            int cb = part * 16;
#pragma unroll
            for (int j = 0; j < 16; ++j)
                orow[j] = (yr[j] - mu) * inv * gamma[cb + j] + beta[cb + j];
        }
    }
}

// ---------------- launch ----------------
extern "C" void kernel_launch(void* const* d_in, const int* in_sizes, int n_in,
                              void* d_out, int out_size, void* d_ws, size_t ws_size,
                              hipStream_t stream) {
    const float* sf    = (const float*)d_in[0];
    const float* rf    = (const float*)d_in[1];
    const float* ef    = (const float*)d_in[2];
    const int*   snd   = (const int*)d_in[3];
    const int*   rcv   = (const int*)d_in[4];
    const float* W1    = (const float*)d_in[5];
    const float* b1    = (const float*)d_in[6];
    const float* W2    = (const float*)d_in[7];
    const float* b2    = (const float*)d_in[8];
    const float* W3    = (const float*)d_in[9];
    const float* b3    = (const float*)d_in[10];
    const float* gamma = (const float*)d_in[11];
    const float* beta  = (const float*)d_in[12];
    float* outp = (float*)d_out;
    const int E = in_sizes[3];

    unsigned short* W1p = (unsigned short*)d_ws;        // 384*128 bf16
    unsigned short* W2p = W1p + 384 * 128;              // 128*128 bf16
    unsigned short* W3p = W2p + 128 * 128;              // 128*128 bf16

    // prepack weights into WMMA B-fragment order (bf16)
    prepack_w<<<dim3((96 * 32 + 255) / 256), 256, 0, stream>>>(W1, W1p, 384, 128);
    prepack_w<<<dim3((32 * 32 + 255) / 256), 256, 0, stream>>>(W2, W2p, 128, 128);
    prepack_w<<<dim3((32 * 32 + 255) / 256), 256, 0, stream>>>(W3, W3p, 128, 128);

    int nblk = (E + MTILE - 1) / MTILE;
    edge_mlp_kernel<<<nblk, 256, 0, stream>>>(sf, rf, ef, snd, rcv,
                                              W1p, b1, W2p, b2, W3p, b3,
                                              gamma, beta, outp, E);
}
